// ResnetCnnsovnetDynamicRouting_17549236371801
// MI455X (gfx1250) — compile-verified, matrix operand order checked by reading, not run on hardware
//
#include <hip/hip_runtime.h>
#include <hip/hip_bf16.h>
#include <math.h>

// =====================================================================
// ResnetCnnsovnetDynamicRouting forward for MI455X (gfx1250, wave32).
// Heavy contractions (5 capsule convs as implicit GEMM + 3 recon FCs) run
// through v_wmma_f32_16x16x32_f16 with BOTH operands pre-packed into the
// CDNA5 per-lane fragment layout, so the GEMM inner loop is pure
// {b128 loads + wmma}.  One wave computes NT (4 or 2) adjacent N-tiles
// sharing the A fragment.
// Workspace requirement: ~630 MB.
// =====================================================================

typedef __attribute__((ext_vector_type(16))) _Float16 v16h;
typedef __attribute__((ext_vector_type(8)))  float    v8f;

#define LN_EPS 1e-5f
#define BN_EPS 1e-5f

// K offset inside a 32-wide K block for (lane, r, j) per the CDNA5 16-bit
// fragment layout: lanes 0-15 hold K 0..7 / 16..23, lanes 16-31 hold 8..15 / 24..31.
__host__ __device__ __forceinline__ int frag_koff(int lane, int r, int j)
{
    return ((r & 4) << 2) + ((lane >> 4) << 3) + ((r & 3) << 1) + j;
}

// ---------------------------------------------------------------------
// Generic direct conv (NCHW, valid padding) with fused activation.
// act: 0=none, 1=relu, 2=selu
// ---------------------------------------------------------------------
__global__ void conv_direct(const float* __restrict__ x, const float* __restrict__ w,
                            const float* __restrict__ bias, float* __restrict__ y,
                            int Nn, int Cin, int H, int W, int Cout, int k, int stride,
                            int Ho, int Wo, int act)
{
    size_t total = (size_t)Nn * Cout * Ho * Wo;
    for (size_t i = (size_t)blockIdx.x * blockDim.x + threadIdx.x; i < total;
         i += (size_t)gridDim.x * blockDim.x) {
        size_t t = i;
        int ow = (int)(t % Wo); t /= Wo;
        int oh = (int)(t % Ho); t /= Ho;
        int co = (int)(t % Cout); t /= Cout;
        int n  = (int)t;
        float acc = bias[co];
        const float* wr = w + (size_t)co * Cin * k * k;
        for (int ci = 0; ci < Cin; ++ci) {
            const float* xr = x + (((size_t)n * Cin + ci) * H + (size_t)oh * stride) * W
                                + (size_t)ow * stride;
            for (int kh = 0; kh < k; ++kh)
                for (int kw = 0; kw < k; ++kw)
                    acc += xr[(size_t)kh * W + kw] * wr[(ci * k + kh) * k + kw];
        }
        if (act == 1) acc = fmaxf(acc, 0.f);
        else if (act == 2) {
            const float alpha = 1.6732632423543772f, sc = 1.0507009873554805f;
            acc = sc * (acc > 0.f ? acc : alpha * (__expf(acc) - 1.f));
        }
        y[i] = acc;
    }
}

// ---------------------------------------------------------------------
// In-place LayerNorm over (C,H,W) per sample, per-element gamma/beta.
// ---------------------------------------------------------------------
__global__ void layernorm_inplace(float* __restrict__ x, const float* __restrict__ g,
                                  const float* __restrict__ b, int CHW)
{
    __shared__ float shs[256], shq[256];
    float* xp = x + (size_t)blockIdx.x * CHW;
    float s = 0.f, q = 0.f;
    for (int i = threadIdx.x; i < CHW; i += blockDim.x) { float v = xp[i]; s += v; q += v * v; }
    shs[threadIdx.x] = s; shq[threadIdx.x] = q;
    __syncthreads();
    for (int off = blockDim.x >> 1; off > 0; off >>= 1) {
        if (threadIdx.x < off) { shs[threadIdx.x] += shs[threadIdx.x + off];
                                 shq[threadIdx.x] += shq[threadIdx.x + off]; }
        __syncthreads();
    }
    float mean = shs[0] / CHW;
    float var  = shq[0] / CHW - mean * mean;
    float inv  = rsqrtf(var + LN_EPS);
    for (int i = threadIdx.x; i < CHW; i += blockDim.x)
        xp[i] = (xp[i] - mean) * inv * g[i] + b[i];
}

// ---------------------------------------------------------------------
// im2col straight into A-fragment layout:
//   Af[((mt*kblocks + kb)*32 + lane)*16 + 2r + j] = x(m = mt*16+lane%16,
//                                                     k = kb*32+frag_koff)
// Zero-padded for k >= K.  Requires M % 16 == 0.
// ---------------------------------------------------------------------
__global__ void im2col_frag_f16(const float* __restrict__ x, _Float16* __restrict__ Af,
                                int Nn, int Cin, int H, int W, int k, int stride,
                                int Ho, int Wo, int K, int kblocks)
{
    size_t M = (size_t)Nn * Ho * Wo;
    size_t total = (M >> 4) * (size_t)kblocks * 512;
    for (size_t i = (size_t)blockIdx.x * blockDim.x + threadIdx.x; i < total;
         i += (size_t)gridDim.x * blockDim.x) {
        int within = (int)(i & 511);
        size_t frag = i >> 9;
        int lane = within >> 4, idx = within & 15;
        int r = idx >> 1, j = idx & 1;
        int kb    = (int)(frag % kblocks);
        size_t mt = frag / kblocks;
        int    kk = (kb << 5) + frag_koff(lane, r, j);
        size_t m  = (mt << 4) + (lane & 15);
        _Float16 v = (_Float16)0.f;
        if (kk < K) {
            int s  = (int)(m % (Ho * Wo));
            int n  = (int)(m / (Ho * Wo));
            int oh = s / Wo, ow = s % Wo;
            int ci = kk / (k * k), rem = kk % (k * k);
            int kh = rem / k, kw = rem % k;
            v = (_Float16)x[(((size_t)n * Cin + ci) * H + (size_t)oh * stride + kh) * W
                            + (size_t)ow * stride + kw];
        }
        Af[i] = v;
    }
}

// ---------------------------------------------------------------------
// Generic row-major [M][K] f32 -> A-fragment f16 (for the FC inputs).
// ---------------------------------------------------------------------
__global__ void pack_A_frag_f16(const float* __restrict__ src, _Float16* __restrict__ Af,
                                int M, int K, int kblocks)
{
    size_t total = ((size_t)M >> 4) * kblocks * 512;
    for (size_t i = (size_t)blockIdx.x * blockDim.x + threadIdx.x; i < total;
         i += (size_t)gridDim.x * blockDim.x) {
        int within = (int)(i & 511);
        size_t frag = i >> 9;
        int lane = within >> 4, idx = within & 15;
        int r = idx >> 1, j = idx & 1;
        int kb    = (int)(frag % kblocks);
        size_t mt = frag / kblocks;
        int    kk = (kb << 5) + frag_koff(lane, r, j);
        size_t m  = (mt << 4) + (lane & 15);
        Af[i] = (kk < K) ? (_Float16)src[m * K + kk] : (_Float16)0.f;
    }
}

// ---------------------------------------------------------------------
// Weights -> B-fragment layout: Bf[((kb*tilesN + nt)*32 + lane)*16 + 2r + j]
//   = w[n*sn + k*sk]   (n = nt*16+lane%16, k = kb*32+frag_koff)
// Conv weights (Cout,K):   sn=K,  sk=1
// Row-major FC  (K,N):     sn=1,  sk=N
// ---------------------------------------------------------------------
__global__ void pack_B_frag_f16(const float* __restrict__ w, _Float16* __restrict__ Bf,
                                int N, int K, int kblocks, int sn, int sk)
{
    int tilesN = N >> 4;
    size_t total = (size_t)kblocks * tilesN * 512;
    for (size_t i = (size_t)blockIdx.x * blockDim.x + threadIdx.x; i < total;
         i += (size_t)gridDim.x * blockDim.x) {
        int within = (int)(i & 511);
        size_t frag = i >> 9;
        int lane = within >> 4, idx = within & 15;
        int r = idx >> 1, j = idx & 1;
        int nt = (int)(frag % tilesN);
        int kb = (int)(frag / tilesN);
        int kk = (kb << 5) + frag_koff(lane, r, j);
        int n  = (nt << 4) + (lane & 15);
        Bf[i] = (kk < K) ? (_Float16)w[(size_t)n * sn + (size_t)kk * sk] : (_Float16)0.f;
    }
}

// ---------------------------------------------------------------------
// Fragment-layout WMMA GEMM.  One wave -> NT adjacent 16x16 tiles sharing
// the A fragment.  Inner loop: 1 v16h load (A) + NT v16h loads (B) + NT wmma.
// Epilogue: +bias[n]; mode 0=none 1=relu 2=sigmoid.
// ---------------------------------------------------------------------
template <int NT>
__global__ void gemm_wmma_frag(const _Float16* __restrict__ Af, const _Float16* __restrict__ Bf,
                               float* __restrict__ C, const float* __restrict__ bias,
                               int M, int N, int kblocks, int mode)
{
    int tilesN  = N >> 4;
    int groupsN = tilesN / NT;
    int wave = blockIdx.x * (blockDim.x >> 5) + (threadIdx.x >> 5);
    int mt   = wave / groupsN;
    int ng   = wave - mt * groupsN;
    if ((mt << 4) >= M) return;            // wave-uniform: EXEC stays all-ones
    int lane = threadIdx.x & 31;

    const v16h* ap = (const v16h*)Af + (size_t)mt * kblocks * 32 + lane;
    const v16h* bp = (const v16h*)Bf + (size_t)ng * NT * 32 + lane;

    v8f acc[NT] = {};
    for (int kb = 0; kb < kblocks; ++kb) {
        __builtin_prefetch((const _Float16*)(ap + (size_t)(kb + 4) * 32), 0, 1);
        v16h a = ap[(size_t)kb * 32];
#pragma unroll
        for (int t = 0; t < NT; ++t) {
            v16h b = bp[((size_t)kb * tilesN + t) * 32];
            acc[t] = __builtin_amdgcn_wmma_f32_16x16x32_f16(
                false, a, false, b, (short)0, acc[t], false, false);
        }
    }

    int half = lane >> 4, l16 = lane & 15;
#pragma unroll
    for (int t = 0; t < NT; ++t) {
        int n = ((ng * NT + t) << 4) + l16;
        float bv = bias ? bias[n] : 0.f;
#pragma unroll
        for (int r = 0; r < 8; ++r) {
            int m = (mt << 4) + r + (half << 3);
            float v = acc[t][r] + bv;
            if (mode == 1)      v = fmaxf(v, 0.f);
            else if (mode == 2) v = 1.f / (1.f + __expf(-v));
            C[(size_t)m * N + n] = v;
        }
    }
}

// ---------------------------------------------------------------------
// BatchNorm stats per channel (column of pred[M][C]); stores mean, invstd.
// ---------------------------------------------------------------------
__global__ void bn_stats(const float* __restrict__ pred, float* __restrict__ mv,
                         int M, int C)
{
    __shared__ float shs[256], shq[256];
    int c = blockIdx.x;
    float s = 0.f, q = 0.f;
    for (int i = threadIdx.x; i < M; i += blockDim.x) {
        float v = pred[(size_t)i * C + c]; s += v; q += v * v;
    }
    shs[threadIdx.x] = s; shq[threadIdx.x] = q;
    __syncthreads();
    for (int off = blockDim.x >> 1; off > 0; off >>= 1) {
        if (threadIdx.x < off) { shs[threadIdx.x] += shs[threadIdx.x + off];
                                 shq[threadIdx.x] += shq[threadIdx.x + off]; }
        __syncthreads();
    }
    if (threadIdx.x == 0) {
        float mean = shs[0] / M;
        float var  = shq[0] / M - mean * mean;
        mv[c]     = mean;
        mv[C + c] = rsqrtf(var + BN_EPS);
    }
}

// ---------------------------------------------------------------------
// Dynamic routing (3 iters) with BatchNorm fused into the pred loads.
// One block of IC*OC threads per (b, spatial).  Thread t: oc=t/IC, ic=t%IC.
// caps_out layout: ((b*OC+oc)*16 + d) * HoWo + s   (NCHW for next stage).
// ---------------------------------------------------------------------
__global__ void routing_kernel(const float* __restrict__ pred,
                               const float* __restrict__ mean,
                               const float* __restrict__ invstd,
                               const float* __restrict__ g,
                               const float* __restrict__ bt,
                               float* __restrict__ caps_out,
                               float* __restrict__ ent_out,
                               int IC, int OC, int HoWo, float entScale)
{
    __shared__ float sh[64];
    int pos = blockIdx.x;
    int b = pos / HoWo, s = pos % HoWo;
    int t = threadIdx.x;
    int oc = t / IC, ic = t % IC;
    int C = OC * 16;

    float p[16];
    const float* base = pred + ((size_t)(b * IC + ic) * HoWo + s) * C + oc * 16;
#pragma unroll
    for (int d = 0; d < 16; ++d) {
        int c = oc * 16 + d;
        p[d] = (base[d] - mean[c]) * invstd[c] * g[c] + bt[c];
    }

    float blogit = 0.f, c_coef = 0.f;
    float v[16];
    for (int it = 0; it < 3; ++it) {
        sh[t] = __expf(blogit);
        __syncthreads();
        float denom = 0.f;
        for (int o = 0; o < OC; ++o) denom += sh[o * IC + ic];
        c_coef = sh[t] / denom;
        __syncthreads();

        float nrm2 = 0.f;
#pragma unroll
        for (int d = 0; d < 16; ++d) {
            float xv = c_coef * p[d];
            xv += __shfl_xor(xv, 1, 32);
            xv += __shfl_xor(xv, 2, 32);
            xv += __shfl_xor(xv, 4, 32);   // all 8 lanes of the ic-group hold sum
            v[d] = xv;
            nrm2 += xv * xv;
        }
        float nrm = sqrtf(nrm2);
        float scale = nrm2 / (1.f + nrm2) / (nrm + 1e-8f);
#pragma unroll
        for (int d = 0; d < 16; ++d) v[d] *= scale;
        if (it < 2) {
            float dot = 0.f;
#pragma unroll
            for (int d = 0; d < 16; ++d) dot += p[d] * v[d];
            blogit += dot;
        }
    }

    if (ic == 0) {
        float* op = caps_out + ((size_t)(b * OC + oc) * 16) * HoWo + s;
#pragma unroll
        for (int d = 0; d < 16; ++d) op[(size_t)d * HoWo] = v[d];
    }

    float e = -c_coef * __logf(fmaxf(c_coef, 1e-30f));
    sh[t] = e;
    __syncthreads();
    if (t == 0) {
        float tot = 0.f;
        for (int i = 0; i < IC * OC; ++i) tot += sh[i];
        atomicAdd(ent_out, tot * entScale);
    }
}

// ---------------------------------------------------------------------
// Classification head: norms, softmax class_pred, masked hc (B,32).
// ---------------------------------------------------------------------
__global__ void class_head(const float* __restrict__ caps, float* __restrict__ class_pred,
                           float* __restrict__ hc, int Bn)
{
    int b = blockIdx.x * blockDim.x + threadIdx.x;
    if (b >= Bn) return;
    const float* cb = caps + (size_t)b * 32;
    float n0 = 0.f, n1 = 0.f;
    for (int d = 0; d < 16; ++d) { n0 += cb[d] * cb[d]; n1 += cb[16 + d] * cb[16 + d]; }
    n0 = sqrtf(n0); n1 = sqrtf(n1);
    float mx = fmaxf(n0, n1);
    float e0 = __expf(n0 - mx), e1 = __expf(n1 - mx);
    float inv = 1.f / (e0 + e1);
    class_pred[b * 2]     = e0 * inv;
    class_pred[b * 2 + 1] = e1 * inv;
    int arg = (n1 > n0) ? 1 : 0;   // argmax, ties -> 0 (matches jnp.argmax)
    for (int j = 0; j < 32; ++j)
        hc[(size_t)b * 32 + j] = cb[j] * (((j >> 4) == arg) ? 1.f : 0.f);
}

__global__ void zero_one(float* p) { if (threadIdx.x == 0 && blockIdx.x == 0) *p = 0.f; }

// =====================================================================
// Host-side launch
// =====================================================================
static inline unsigned blks(size_t total, int tpb) {
    size_t b = (total + tpb - 1) / tpb;
    return (unsigned)(b > (1u << 30) ? (1u << 30) : b);
}

static void launch_gemm(const _Float16* Af, const _Float16* Bf, float* C, const float* bias,
                        int M, int N, int kblocks, int mode, hipStream_t stream)
{
    int tilesN = N >> 4;
    int NT = (tilesN % 4 == 0) ? 4 : ((tilesN % 2 == 0) ? 2 : 1);
    size_t waves = (size_t)(M >> 4) * (tilesN / NT);
    unsigned nb = blks(waves * 32, 256);
    if (NT == 4)
        gemm_wmma_frag<4><<<nb, 256, 0, stream>>>(Af, Bf, C, bias, M, N, kblocks, mode);
    else if (NT == 2)
        gemm_wmma_frag<2><<<nb, 256, 0, stream>>>(Af, Bf, C, bias, M, N, kblocks, mode);
    else
        gemm_wmma_frag<1><<<nb, 256, 0, stream>>>(Af, Bf, C, bias, M, N, kblocks, mode);
}

extern "C" void kernel_launch(void* const* d_in, const int* in_sizes, int n_in,
                              void* d_out, int out_size, void* d_ws, size_t ws_size,
                              hipStream_t stream)
{
    // ---- input map (setup_inputs dict insertion order) ----
    const float* x       = (const float*)d_in[0];    // (64,3,128,128)
    const float* pre_w1  = (const float*)d_in[1];
    const float* pre_b1  = (const float*)d_in[2];
    const float* ln1_g   = (const float*)d_in[3];
    const float* ln1_b   = (const float*)d_in[4];
    const float* pre_w2  = (const float*)d_in[5];
    const float* pre_b2  = (const float*)d_in[6];
    const float* ln2_g   = (const float*)d_in[7];
    const float* ln2_b   = (const float*)d_in[8];
    const float* prim_w  = (const float*)d_in[9];
    const float* prim_b  = (const float*)d_in[10];
    const float* lnp_g   = (const float*)d_in[11];
    const float* lnp_b   = (const float*)d_in[12];
    const float* ccw[5], *ccb[5], *ccg[5], *ccbt[5];
    for (int i = 0; i < 5; ++i) {
        ccw[i]  = (const float*)d_in[13 + 4 * i];
        ccb[i]  = (const float*)d_in[14 + 4 * i];
        ccg[i]  = (const float*)d_in[15 + 4 * i];
        ccbt[i] = (const float*)d_in[16 + 4 * i];
    }
    const float* rw1 = (const float*)d_in[33];
    const float* rb1 = (const float*)d_in[34];
    const float* rw2 = (const float*)d_in[35];
    const float* rb2 = (const float*)d_in[36];
    const float* rw3 = (const float*)d_in[37];
    const float* rb3 = (const float*)d_in[38];

    float* out       = (float*)d_out;
    float* out_class = out;          // 128
    float* out_ent   = out + 128;    // 1
    float* out_recon = out + 129;    // 64*3*128*128

    // ---- workspace carve-up (floats, 32B aligned); total ~630 MB ----
    float* ws = (float*)d_ws;
    size_t ofs = 0;
    auto alloc = [&](size_t nfl) {
        float* p = ws + ofs; ofs += (nfl + 7) & ~(size_t)7; return p;
    };
    float* A1    = alloc((size_t)64 * 32 * 126 * 126);    // conv1/LN1
    float* A2    = alloc((size_t)64 * 32 * 62 * 62);      // conv2/LN2
    float* A3    = alloc((size_t)64 * 128 * 30 * 30);     // primary caps (512,16,30,30)
    float* capsA = alloc((size_t)512 * 16 * 28 * 28);
    float* capsB = alloc((size_t)512 * 16 * 28 * 28);
    float* pred  = alloc((size_t)401408 * 128);           // max M*Cout
    float* mv    = alloc(256);
    float* hc    = alloc(64 * 32);
    float* h1    = alloc(64 * 256);
    float* h2    = alloc(64 * 512);
    _Float16* colbuf = (_Float16*)alloc(32112640);        // 64.2M f16 (max M*Kpad)
    _Float16* wB     = (_Float16*)alloc(12582912);        // 25.1M f16 (max Kpad*N)
    (void)ws_size; (void)in_sizes; (void)n_in; (void)out_size;

    // ---- pre layers ----
    conv_direct<<<blks((size_t)64 * 32 * 126 * 126, 256), 256, 0, stream>>>(
        x, pre_w1, pre_b1, A1, 64, 3, 128, 128, 32, 3, 1, 126, 126, /*relu*/1);
    layernorm_inplace<<<64, 256, 0, stream>>>(A1, ln1_g, ln1_b, 32 * 126 * 126);

    conv_direct<<<blks((size_t)64 * 32 * 62 * 62, 256), 256, 0, stream>>>(
        A1, pre_w2, pre_b2, A2, 64, 32, 126, 126, 32, 3, 2, 62, 62, /*relu*/1);
    layernorm_inplace<<<64, 256, 0, stream>>>(A2, ln2_g, ln2_b, 32 * 62 * 62);

    conv_direct<<<blks((size_t)64 * 128 * 30 * 30, 256), 256, 0, stream>>>(
        A2, prim_w, prim_b, A3, 64, 32, 62, 62, 128, 3, 2, 30, 30, /*selu*/2);
    layernorm_inplace<<<64, 256, 0, stream>>>(A3, lnp_g, lnp_b, 128 * 30 * 30);

    zero_one<<<1, 1, 0, stream>>>(out_ent);

    // ---- capsule stages ----
    struct Stage { int Hin, k, stride, Ho, OC; };
    const Stage st[5] = { {30,3,1,28,8}, {28,3,2,13,8}, {13,5,1,9,8}, {9,3,2,4,8}, {4,4,1,1,2} };
    const float* cin = A3;
    float* cout_bufs[2] = { capsA, capsB };
    for (int i = 0; i < 5; ++i) {
        const int IC = 8, Cin = 16;
        int k = st[i].k, Ho = st[i].Ho, OC = st[i].OC;
        int Cout = OC * 16;
        int K = Cin * k * k;
        int kblocks = (K + 31) >> 5;
        int HoWo = Ho * Ho;
        size_t M = (size_t)512 * HoWo;
        float* cap_out = cout_bufs[i & 1];

        pack_B_frag_f16<<<blks((size_t)kblocks * (Cout >> 4) * 512, 256), 256, 0, stream>>>(
            ccw[i], wB, Cout, K, kblocks, /*sn=*/K, /*sk=*/1);
        im2col_frag_f16<<<blks((M >> 4) * kblocks * 512, 256), 256, 0, stream>>>(
            cin, colbuf, 512, Cin, st[i].Hin, st[i].Hin, k, st[i].stride, Ho, Ho, K, kblocks);

        launch_gemm(colbuf, wB, pred, ccb[i], (int)M, Cout, kblocks, /*mode*/0, stream);

        bn_stats<<<Cout, 256, 0, stream>>>(pred, mv, (int)M, Cout);

        float entScale = 1.0f / (logf((float)OC) * IC * HoWo);
        routing_kernel<<<64 * HoWo, IC * OC, 0, stream>>>(
            pred, mv, mv + Cout, ccg[i], ccbt[i], cap_out, out_ent, IC, OC, HoWo, entScale);
        cin = cap_out;
    }

    // ---- classification + reconstruction ----
    class_head<<<1, 64, 0, stream>>>(cin, out_class, hc, 64);

    // FC1: (64,32)x(32,256) relu
    pack_A_frag_f16<<<blks((size_t)4 * 1 * 512, 256), 256, 0, stream>>>(hc, colbuf, 64, 32, 1);
    pack_B_frag_f16<<<blks((size_t)1 * 16 * 512, 256), 256, 0, stream>>>(
        rw1, wB, 256, 32, 1, /*sn=*/1, /*sk=*/256);
    launch_gemm(colbuf, wB, h1, rb1, 64, 256, 1, /*relu*/1, stream);

    // FC2: (64,256)x(256,512) relu
    pack_A_frag_f16<<<blks((size_t)4 * 8 * 512, 256), 256, 0, stream>>>(h1, colbuf, 64, 256, 8);
    pack_B_frag_f16<<<blks((size_t)8 * 32 * 512, 256), 256, 0, stream>>>(
        rw2, wB, 512, 256, 8, /*sn=*/1, /*sk=*/512);
    launch_gemm(colbuf, wB, h2, rb2, 64, 512, 8, /*relu*/1, stream);

    // FC3: (64,512)x(512,49152) sigmoid -> recon directly into d_out
    pack_A_frag_f16<<<blks((size_t)4 * 16 * 512, 256), 256, 0, stream>>>(h2, colbuf, 64, 512, 16);
    pack_B_frag_f16<<<blks((size_t)16 * 3072 * 512, 256), 256, 0, stream>>>(
        rw3, wB, 49152, 512, 16, /*sn=*/1, /*sk=*/49152);
    launch_gemm(colbuf, wB, out_recon, rb3, 64, 49152, 16, /*sigmoid*/2, stream);
}
